// TransformerConvPoolingBlock_33500744909410
// MI455X (gfx1250) — matrix operand order repaired and checked by reading, not compiled, not run on hardware
//
#include <hip/hip_runtime.h>
#include <hip/hip_bf16.h>

typedef __attribute__((ext_vector_type(16))) _Float16 v16h;
typedef __attribute__((ext_vector_type(8)))  _Float16 v8h;
typedef __attribute__((ext_vector_type(8)))  float    v8f;

// ---------------------------------------------------------------------------
// helpers
// ---------------------------------------------------------------------------
static __device__ __forceinline__ float atomicMaxFloat(float* addr, float val) {
  // IEEE ordered-int trick; works for mixed signs and -inf init.
  if (val >= 0.0f)
    return __int_as_float(atomicMax((int*)addr, __float_as_int(val)));
  else
    return __uint_as_float(atomicMin((unsigned int*)addr, __float_as_uint(val)));
}

__global__ void fill_f32_kernel(float* __restrict__ p, float val, int n) {
  int i = blockIdx.x * blockDim.x + threadIdx.x;
  if (i < n) p[i] = val;
}

__global__ void f32_to_f16_kernel(const float* __restrict__ in,
                                  _Float16* __restrict__ out, int n) {
  int i = blockIdx.x * blockDim.x + threadIdx.x;
  if (i < n) out[i] = (_Float16)in[i];
}

// W [K, Nc] f32 row-major  ->  Wt [Nc, K] f16 row-major (transposed pack)
__global__ void transpose_pack_f16_kernel(const float* __restrict__ W,
                                          _Float16* __restrict__ Wt,
                                          int K, int Nc) {
  int i = blockIdx.x * blockDim.x + threadIdx.x;
  if (i >= K * Nc) return;
  int n = i / K;       // output row (coalesced writes)
  int kk = i - n * K;  // output col
  Wt[(size_t)n * K + kk] = (_Float16)W[(size_t)kk * Nc + n];
}

// ---------------------------------------------------------------------------
// WMMA GEMM:  C[M,Nc] = A[M,K](f16) * Bt[Nc,K](f16, pre-transposed) + bias[Nc]
// Each wave computes a 16x64 output strip: the A fragment is loaded once per
// 32-deep k-step and reused by 4 independent WMMA accumulators (4x arithmetic
// intensity on the A stream, 4 co-issuable v_wmma per k-step).
// K is compile-time -> fully unrolled: 16 (K=128) / 32 (K=256) static wmma.
// A/B fragments use contiguous k order (k = 16*half + s), same in both (wave32).
// ---------------------------------------------------------------------------
template <int K>
__global__ void wmma_gemm_kernel(const _Float16* __restrict__ A,
                                 const _Float16* __restrict__ Bt,
                                 const float* __restrict__ bias,
                                 float* __restrict__ C,
                                 int M, int Nc, int relu) {
  const int wave = blockIdx.x * (blockDim.x >> 5) + (threadIdx.x >> 5);
  const int lane = threadIdx.x & 31;
  const int groupsN = Nc >> 6;  // 64-wide N strips
  const int total = (M >> 4) * groupsN;
  if (wave >= total) return;

  const int tm = wave / groupsN;
  const int gn = wave - tm * groupsN;
  const int m0 = tm << 4, n0 = gn << 6;
  const int lr = lane & 15;   // row (A) / col (B) within tile
  const int hi = lane >> 4;   // half-wave select

  // A: lane<16 holds K {0..15}+k0 of row m0+lr; lane>=16 holds {16..31}
  const _Float16* Abase = A + (size_t)(m0 + lr) * K + (hi ? 16 : 0);
  // B: lane<16 holds K {0..15}+k0 of col n0+lr; lane>=16 holds {16..31}
  const _Float16* Bbase = Bt + (size_t)(n0 + lr) * K + (hi ? 16 : 0);
  const size_t bstride = (size_t)16 * K;  // next 16-col tile of Bt

  v8f acc0 = {}, acc1 = {}, acc2 = {}, acc3 = {};
#pragma unroll
  for (int k0 = 0; k0 < K; k0 += 32) {
    v8h a0 = *(const v8h*)(Abase + k0);
    v8h a1 = *(const v8h*)(Abase + k0 + 8);
    v16h a = __builtin_shufflevector(a0, a1, 0, 1, 2, 3, 4, 5, 6, 7,
                                     8, 9, 10, 11, 12, 13, 14, 15);
    v16h b0 = *(const v16h*)(Bbase + k0);
    v16h b1 = *(const v16h*)(Bbase + bstride + k0);
    v16h b2 = *(const v16h*)(Bbase + 2 * bstride + k0);
    v16h b3 = *(const v16h*)(Bbase + 3 * bstride + k0);
    acc0 = __builtin_amdgcn_wmma_f32_16x16x32_f16(false, a, false, b0,
                                                  (short)0, acc0, false, false);
    acc1 = __builtin_amdgcn_wmma_f32_16x16x32_f16(false, a, false, b1,
                                                  (short)0, acc1, false, false);
    acc2 = __builtin_amdgcn_wmma_f32_16x16x32_f16(false, a, false, b2,
                                                  (short)0, acc2, false, false);
    acc3 = __builtin_amdgcn_wmma_f32_16x16x32_f16(false, a, false, b3,
                                                  (short)0, acc3, false, false);
  }

  // Epilogue: C/D layout — lane gives column (n0 + j*16 + lr), VGPR r gives
  // row m0 + r (+8 for the upper half-wave).
  float* Crow = C + (size_t)(m0 + hi * 8) * Nc;
#pragma unroll
  for (int j = 0; j < 4; ++j) {
    const v8f acc = (j == 0) ? acc0 : (j == 1) ? acc1 : (j == 2) ? acc2 : acc3;
    const int n = n0 + j * 16 + lr;
    const float bsv = bias[n];
#pragma unroll
    for (int r = 0; r < 8; ++r) {
      float val = acc[r] + bsv;
      if (relu) val = fmaxf(val, 0.0f);
      Crow[(size_t)r * Nc + n] = val;
    }
  }
}

// ---------------------------------------------------------------------------
// Edge attention kernels.  q,k,v are [N, 256] f32 (H=2 heads x D=128).
// ---------------------------------------------------------------------------
__global__ void edge_logits_max_kernel(const long long* __restrict__ src,
                                       const long long* __restrict__ dst,
                                       const float* __restrict__ q,
                                       const float* __restrict__ k,
                                       float* __restrict__ logits,   // [E,2]
                                       float* __restrict__ mbuf,     // [N,2]
                                       int E_) {
  const int wave = blockIdx.x * (blockDim.x >> 5) + (threadIdx.x >> 5);
  const int lane = threadIdx.x & 31;
  if (wave >= E_) return;
  const int s = (int)src[wave], d = (int)dst[wave];
  // lanes 0..15 cover features 0..127 (head 0), lanes 16..31 -> head 1
  const float4* qp = (const float4*)(q + (size_t)d * 256) + lane * 2;
  const float4* kp = (const float4*)(k + (size_t)s * 256) + lane * 2;
  float4 q0 = qp[0], q1 = qp[1], k0 = kp[0], k1 = kp[1];
  float part = q0.x * k0.x + q0.y * k0.y + q0.z * k0.z + q0.w * k0.w +
               q1.x * k1.x + q1.y * k1.y + q1.z * k1.z + q1.w * k1.w;
  // reduce within each 16-lane half (wave32)
#pragma unroll
  for (int off = 1; off < 16; off <<= 1) part += __shfl_xor(part, off, 32);
  if ((lane & 15) == 0) {
    const int h = lane >> 4;
    const float lg = part * 0.08838834764831845f;  // 1/sqrt(128)
    logits[(size_t)wave * 2 + h] = lg;
    atomicMaxFloat(&mbuf[(size_t)d * 2 + h], lg);
  }
}

__global__ void edge_p_denom_kernel(const long long* __restrict__ dst,
                                    const float* __restrict__ logits,
                                    const float* __restrict__ mbuf,
                                    float* __restrict__ pbuf,    // [E,2]
                                    float* __restrict__ denom,   // [N,2]
                                    int E_) {
  int i = blockIdx.x * blockDim.x + threadIdx.x;
  if (i >= E_ * 2) return;
  const int e = i >> 1, h = i & 1;
  const int d = (int)dst[e];
  const float p = __expf(logits[i] - mbuf[(size_t)d * 2 + h]);
  pbuf[i] = p;
  atomicAdd(&denom[(size_t)d * 2 + h], p);
}

__global__ void edge_aggregate_kernel(const long long* __restrict__ src,
                                      const long long* __restrict__ dst,
                                      const float* __restrict__ pbuf,
                                      const float* __restrict__ denom,
                                      const float* __restrict__ v,
                                      float* __restrict__ accum,  // [N,256]
                                      int E_) {
  const int wave = blockIdx.x * (blockDim.x >> 5) + (threadIdx.x >> 5);
  const int lane = threadIdx.x & 31;
  if (wave >= E_) return;
  const int s = (int)src[wave], d = (int)dst[wave];
  const int h = lane >> 4;  // lanes 0-15: head0 feats, 16-31: head1 feats
  const float alpha =
      pbuf[(size_t)wave * 2 + h] / (denom[(size_t)d * 2 + h] + 1e-16f);
  const float* vp = v + (size_t)s * 256 + lane * 8;
  float* op = accum + (size_t)d * 256 + lane * 8;
  float4 v0 = *(const float4*)vp;
  float4 v1 = *(const float4*)(vp + 4);
  atomicAdd(op + 0, alpha * v0.x);
  atomicAdd(op + 1, alpha * v0.y);
  atomicAdd(op + 2, alpha * v0.z);
  atomicAdd(op + 3, alpha * v0.w);
  atomicAdd(op + 4, alpha * v1.x);
  atomicAdd(op + 5, alpha * v1.y);
  atomicAdd(op + 6, alpha * v1.z);
  atomicAdd(op + 7, alpha * v1.w);
}

// ---------------------------------------------------------------------------
// Pooling
// ---------------------------------------------------------------------------
__global__ void pool_kernel(const float* __restrict__ xf,        // [N,128]
                            const long long* __restrict__ gi,    // [N]
                            float* __restrict__ xmax,            // [G,128]
                            float* __restrict__ xsum,            // [G,128]
                            float* __restrict__ cnt,             // [G]
                            int Nn) {
  int i = blockIdx.x * blockDim.x + threadIdx.x;
  if (i >= Nn * 128) return;
  const int node = i >> 7, d = i & 127;
  const int g = (int)gi[node];
  const float val = xf[i];
  atomicMaxFloat(&xmax[(size_t)g * 128 + d], val);
  atomicAdd(&xsum[(size_t)g * 128 + d], val);
  if (d == 0) atomicAdd(&cnt[g], 1.0f);
}

__global__ void finalize_kernel(const float* __restrict__ xmax,
                                const float* __restrict__ xsum,
                                const float* __restrict__ cnt,
                                float* __restrict__ out, int G_) {
  int i = blockIdx.x * blockDim.x + threadIdx.x;
  if (i >= G_ * 384) return;
  const int g = i / 384, c = i - g * 384;
  float val;
  if (c < 128)
    val = xmax[(size_t)g * 128 + c];
  else if (c < 256)
    val = xsum[(size_t)g * 128 + (c - 128)] / fmaxf(cnt[g], 1.0f);
  else
    val = xsum[(size_t)g * 128 + (c - 256)];
  out[i] = val;
}

// ---------------------------------------------------------------------------
// host launcher
// ---------------------------------------------------------------------------
extern "C" void kernel_launch(void* const* d_in, const int* in_sizes, int n_in,
                              void* d_out, int out_size, void* d_ws,
                              size_t ws_size, hipStream_t stream) {
  (void)in_sizes; (void)n_in; (void)out_size;
  const int N = 50000, E = 800000, D = 128, HD = 256, Lnum = 2, G = 64;

  const float* x_in = (const float*)d_in[0];
  const long long* edge = (const long long*)d_in[1];
  const long long* gi = (const long long*)d_in[2];
  const float* Wq = (const float*)d_in[3];
  const float* bq = (const float*)d_in[4];
  const float* Wk = (const float*)d_in[5];
  const float* bk = (const float*)d_in[6];
  const float* Wv = (const float*)d_in[7];
  const float* bv = (const float*)d_in[8];
  const float* Wsk = (const float*)d_in[9];
  const float* bsk = (const float*)d_in[10];
  const float* Wt = (const float*)d_in[11];
  const float* bt = (const float*)d_in[12];
  float* out = (float*)d_out;

  const long long* src = edge;
  const long long* dst = edge + E;

  // workspace layout
  char* ws = (char*)d_ws;
  size_t off = 0;
  auto alloc = [&](size_t bytes) -> void* {
    void* p = ws + off;
    off = (off + bytes + 255) & ~(size_t)255;
    return p;
  };
  _Float16* hbuf  = (_Float16*)alloc((size_t)N * HD * sizeof(_Float16)); // f16 A (reused)
  float* qb    = (float*)alloc((size_t)N * HD * sizeof(float));
  float* kb    = (float*)alloc((size_t)N * HD * sizeof(float));
  float* vb    = (float*)alloc((size_t)N * HD * sizeof(float));
  float* accum = (float*)alloc((size_t)N * HD * sizeof(float));
  float* xbuf  = (float*)alloc((size_t)N * D * sizeof(float));
  float* logits = (float*)alloc((size_t)E * 2 * sizeof(float));
  float* pbuf   = (float*)alloc((size_t)E * 2 * sizeof(float));
  float* mbuf   = (float*)alloc((size_t)N * 2 * sizeof(float));
  float* denom  = (float*)alloc((size_t)N * 2 * sizeof(float));
  _Float16* wqT = (_Float16*)alloc((size_t)HD * D * sizeof(_Float16));
  _Float16* wkT = (_Float16*)alloc((size_t)HD * D * sizeof(_Float16));
  _Float16* wvT = (_Float16*)alloc((size_t)HD * D * sizeof(_Float16));
  _Float16* wsT = (_Float16*)alloc((size_t)HD * D * sizeof(_Float16));
  _Float16* wtT = (_Float16*)alloc((size_t)D * HD * sizeof(_Float16));
  float* xmax = (float*)alloc((size_t)G * D * sizeof(float));
  float* xsum = (float*)alloc((size_t)G * D * sizeof(float));
  float* cnt  = (float*)alloc((size_t)G * sizeof(float));
  if (off > ws_size) return;  // workspace too small; bail deterministically

  const int TB = 256;
  auto blocks = [](long long n, int tb) { return (unsigned)((n + tb - 1) / tb); };

  // GEMM launch config: one wave per 16x64 strip, 8 waves (256 thr) per block
  const int gemmThreads = 256;
  const int stripsQKV = (N / 16) * (HD / 64);  // 3125 * 4 = 12500 waves
  const int stripsOut = (N / 16) * (D / 64);   // 3125 * 2 = 6250 waves
  const unsigned gemmBlksQKV = blocks(stripsQKV, gemmThreads / 32);
  const unsigned gemmBlksOut = blocks(stripsOut, gemmThreads / 32);

  const float* xcur = x_in;  // current layer input [N,128] f32

  for (int l = 0; l < Lnum; ++l) {
    // 1) pack activations to f16
    f32_to_f16_kernel<<<blocks((long long)N * D, TB), TB, 0, stream>>>(
        xcur, hbuf, N * D);

    // 2) pack/transpose this layer's weights to f16
    transpose_pack_f16_kernel<<<blocks((long long)D * HD, TB), TB, 0, stream>>>(
        Wq + (size_t)l * D * HD, wqT, D, HD);
    transpose_pack_f16_kernel<<<blocks((long long)D * HD, TB), TB, 0, stream>>>(
        Wk + (size_t)l * D * HD, wkT, D, HD);
    transpose_pack_f16_kernel<<<blocks((long long)D * HD, TB), TB, 0, stream>>>(
        Wv + (size_t)l * D * HD, wvT, D, HD);
    transpose_pack_f16_kernel<<<blocks((long long)D * HD, TB), TB, 0, stream>>>(
        Wsk + (size_t)l * D * HD, wsT, D, HD);
    transpose_pack_f16_kernel<<<blocks((long long)HD * D, TB), TB, 0, stream>>>(
        Wt + (size_t)l * HD * D, wtT, HD, D);

    // 3) dense projections via WMMA (K=128): q, k, v, skip->accum
    wmma_gemm_kernel<128><<<gemmBlksQKV, gemmThreads, 0, stream>>>(
        hbuf, wqT, bq + (size_t)l * HD, qb, N, HD, 0);
    wmma_gemm_kernel<128><<<gemmBlksQKV, gemmThreads, 0, stream>>>(
        hbuf, wkT, bk + (size_t)l * HD, kb, N, HD, 0);
    wmma_gemm_kernel<128><<<gemmBlksQKV, gemmThreads, 0, stream>>>(
        hbuf, wvT, bv + (size_t)l * HD, vb, N, HD, 0);
    wmma_gemm_kernel<128><<<gemmBlksQKV, gemmThreads, 0, stream>>>(
        hbuf, wsT, bsk + (size_t)l * HD, accum, N, HD, 0);

    // 4) segment-softmax attention over edges
    fill_f32_kernel<<<blocks((long long)N * 2, TB), TB, 0, stream>>>(
        mbuf, -__builtin_inff(), N * 2);
    fill_f32_kernel<<<blocks((long long)N * 2, TB), TB, 0, stream>>>(
        denom, 0.0f, N * 2);
    edge_logits_max_kernel<<<blocks((long long)E * 32, TB), TB, 0, stream>>>(
        src, dst, qb, kb, logits, mbuf, E);
    edge_p_denom_kernel<<<blocks((long long)E * 2, TB), TB, 0, stream>>>(
        dst, logits, mbuf, pbuf, denom, E);
    edge_aggregate_kernel<<<blocks((long long)E * 32, TB), TB, 0, stream>>>(
        src, dst, pbuf, denom, vb, accum, E);

    // 5) output transform + ReLU via WMMA (K=256)
    f32_to_f16_kernel<<<blocks((long long)N * HD, TB), TB, 0, stream>>>(
        accum, hbuf, N * HD);
    wmma_gemm_kernel<256><<<gemmBlksOut, gemmThreads, 0, stream>>>(
        hbuf, wtT, bt + (size_t)l * D, xbuf, N, D, 1);

    xcur = xbuf;
  }

  // 6) per-graph max / mean / sum pooling
  fill_f32_kernel<<<blocks((long long)G * D, TB), TB, 0, stream>>>(
      xmax, -__builtin_inff(), G * D);
  fill_f32_kernel<<<blocks((long long)G * D, TB), TB, 0, stream>>>(
      xsum, 0.0f, G * D);
  fill_f32_kernel<<<1, 64, 0, stream>>>(cnt, 0.0f, G);
  pool_kernel<<<blocks((long long)N * D, TB), TB, 0, stream>>>(
      xbuf, gi, xmax, xsum, cnt, N);
  finalize_kernel<<<blocks((long long)G * 384, TB), TB, 0, stream>>>(
      xmax, xsum, cnt, out, G);
}
